// NeuralSplineFlow_39883066310903
// MI455X (gfx1250) — matrix operand order repaired and checked by reading, not compiled, not run on hardware
//
#include <hip/hip_runtime.h>
#include <hip/hip_bf16.h>
#include <math.h>

typedef __attribute__((ext_vector_type(16))) _Float16 v16h;
typedef __attribute__((ext_vector_type(8)))  _Float16 v8h;
typedef __attribute__((ext_vector_type(8)))  float    v8f;
typedef __attribute__((ext_vector_type(4)))  int      v4i;
typedef _Float16 half_t;

/* address-space-qualified pointers for the gfx1250 async LDS builtin:
   signature is (v4i AS1*, v4i AS3*, imm offset, imm cpol) per hipcc diag */
typedef __attribute__((address_space(1))) v4i* gv4i_p;
typedef __attribute__((address_space(3))) v4i* lv4i_p;

#define NL     16
#define NBLK   2
#define HDIM   128
#define PM     23
#define NB     8
#define TBOUND 3.0f
#define MINW   0.001f
#define MIND   0.001f
#define DCONST 0.53971067f      /* log(expm1(1 - MIND)) */
#define SQH    11.313708499f    /* sqrt(128) */
#define LU_EPS 0.001f

#define ROWS 64        /* rows per workgroup */
#define NTHR 128       /* 4 wave32s */
#define AS   136       /* LDS activation stride (f16), padded vs 64 banks */
#define WS   136       /* LDS weight stride (f16) */
#define HS   132       /* LDS h stride (f32) */
#define PS   32        /* LDS spline-param stride (f32) */

#define WBUF_ELEMS (HDIM * WS)      /* one weight buffer, f16 elements */
#define NGEMM_TOTAL (NL * 2 * 5)    /* 160 GEMMs over the whole flow */

/* ---- gfx1250 async LDS path (guarded; falls back to sync copy) ---- */
#if __has_builtin(__builtin_amdgcn_global_load_async_to_lds_b128) && \
    __has_builtin(__builtin_amdgcn_s_wait_asynccnt)
#define HAVE_ASYNC_LDS 1
#endif

__device__ __forceinline__ void async_wait(){
#if defined(HAVE_ASYNC_LDS)
  __builtin_amdgcn_s_wait_asynccnt(0);
#endif
}

/* wait for in-flight weight tile, then rendezvous: one hand-off per GEMM */
#define WAITBAR() do { async_wait(); __syncthreads(); } while (0)

__device__ __forceinline__ float softplusf(float x){
  return (x > 20.f) ? x : log1pf(expf(x));
}

__device__ __forceinline__ v16h ld_frag(const half_t* p0, const half_t* p1){
  union { v16h v; v8h h[2]; } u;
  u.h[0] = *(const v8h*)p0;
  u.h[1] = *(const v8h*)p1;
  return u.v;
}

/* stage a [rows x 128] f16 weight matrix (contiguous in global) into LDS
   (stride WS). NO barriers inside: the WAITBAR() pipeline provides them.
   Async path streams L2->LDS while WMMAs on the other buffer execute. */
__device__ __forceinline__ void stage_w(const half_t* __restrict__ src,
                                        half_t* dst, int rows){
  int chunks = rows * (HDIM / 8);          /* 16-byte chunks */
#if defined(HAVE_ASYNC_LDS)
  for (int c = threadIdx.x; c < chunks; c += NTHR){
    int r = c >> 4, kc = c & 15;
    __builtin_amdgcn_global_load_async_to_lds_b128(
        (gv4i_p)(void*)(src + (size_t)c * 8),
        (lv4i_p)(void*)(dst + r * WS + kc * 8),
        0, 0);
  }
#else
  const uint4* s = (const uint4*)src;
  for (int c = threadIdx.x; c < chunks; c += NTHR){
    int r = c >> 4, kc = c & 15;
    *(uint4*)&dst[r * WS + kc * 8] = s[c];
  }
#endif
}

/* prefetch weights for global GEMM index t into wbuf0 + (t&1)*WBUF_ELEMS.
   Buffer select is pure offset arithmetic on ONE LDS base pointer so the
   accesses provably stay in addrspace(3) (ds_*, not flat_*). */
__device__ __forceinline__ void prefetch_w(int t, const half_t* __restrict__ wres_h,
                                           const half_t* __restrict__ wf_h,
                                           half_t* wbuf0){
  if (t >= NGEMM_TOTAL) return;
  half_t* dst = wbuf0 + (size_t)(t & 1) * WBUF_ELEMS;
  int li = t / 10;
  int g  = t % 5;
  if (g < 4)
    stage_w(wres_h + (((size_t)li * NBLK + (g >> 1)) * 2 + (g & 1)) * HDIM * HDIM,
            dst, HDIM);
  else
    stage_w(wf_h + (size_t)li * 32 * HDIM, dst, 32);
}

/* One wave: D[16 x (NT*16)] = A[16x128] * W^T via v_wmma_f32_16x16x32_f16.
   A (16-bit 16x32): lanes 0-15 row=lane, K={0..7,16..23}; lanes 16-31 K={8..15,24..31}.
   B (K x N): lane n<16 holds col n K={0..7,16..23}; lanes 16-31 K={8..15,24..31}. */
template<int NT>
__device__ __forceinline__ void gemm16(const half_t* Abase, const half_t* wbuf, v8f* acc){
  const int lane = threadIdx.x & 31;
  const int hi   = lane >> 4;
  const int l15  = lane & 15;
  #pragma unroll
  for (int kt = 0; kt < 4; ++kt){
    int ka = kt * 32 + hi * 8;
    v16h a = ld_frag(&Abase[l15 * AS + ka], &Abase[l15 * AS + ka + 16]);
    int kb = kt * 32 + hi * 8;
    #pragma unroll
    for (int nt = 0; nt < NT; ++nt){
      v16h b = ld_frag(&wbuf[(nt * 16 + l15) * WS + kb],
                       &wbuf[(nt * 16 + l15) * WS + kb + 16]);
      acc[nt] = __builtin_amdgcn_wmma_f32_16x16x32_f16(
          false, a, false, b, (short)0, acc[nt], false, false);
    }
  }
}

/* ---------------- weight pre-conversion kernels ---------------- */

__global__ void cvt_wres(const float* __restrict__ src, half_t* __restrict__ dst, int n){
  int i = blockIdx.x * 256 + threadIdx.x;
  if (i < n) dst[i] = (half_t)src[i];
}

__global__ void cvt_wf(const float* __restrict__ wf, half_t* __restrict__ dst){
  int i = blockIdx.x * 256 + threadIdx.x;     /* over NL*32*128, rows 23..31 zero pad */
  if (i >= NL * 32 * HDIM) return;
  int k  = i & 127;
  int n  = (i >> 7) & 31;
  int li = i >> 12;
  dst[i] = (n < PM) ? (half_t)wf[((size_t)li * (2 * PM) + PM + n) * HDIM + k] : (half_t)0.f;
}

/* ---------------- main flow kernel ---------------- */

__global__ __launch_bounds__(NTHR)
void nsf_main(const float* __restrict__ z0g, const float* __restrict__ xg,
              const float* __restrict__ sigmag,
              const float* __restrict__ n1w1, const float* __restrict__ n1b1,
              const float* __restrict__ n1w2, const float* __restrict__ n1b2,
              const float* __restrict__ n2w1, const float* __restrict__ n2b1,
              const float* __restrict__ n2w2, const float* __restrict__ n2b2,
              const float* __restrict__ made_W0, const float* __restrict__ made_b0,
              const half_t* __restrict__ wres_h, const float* __restrict__ made_bres,
              const half_t* __restrict__ wf_h, const float* __restrict__ made_bf,
              const float* __restrict__ lu_lower, const float* __restrict__ lu_upper,
              const float* __restrict__ lu_diag, const float* __restrict__ lu_bias,
              const int* __restrict__ perms, float* __restrict__ outp, int B)
{
  extern __shared__ char smem[];
  half_t* wbuf0 = (half_t*)smem;                 /* 2 x 128*136 f16 (double-buffered) */
  half_t* actA  = wbuf0 + 2 * WBUF_ELEMS;        /* 64*136 f16 */
  float*  hbuf  = (float*)(actA + ROWS * AS);    /* 64*132 f32 */
  float*  pbuf  = hbuf + ROWS * HS;              /* 64*32  f32 */
  float*  zbuf  = pbuf + ROWS * PS;              /* 64*2 */
  float*  outb  = zbuf + ROWS * 2;               /* 64*2 */
  float*  ldb   = outb + ROWS * 2;               /* 64*2 */
  float*  ldtot = ldb + ROWS * 2;                /* 64   */

  const int tid   = threadIdx.x;
  const int wave  = tid >> 5;
  const int lane  = tid & 31;
  const int hi    = lane >> 4;
  const int l15   = lane & 15;
  const int m0    = wave * 16;                   /* wave's private 16-row M-tile */
  const int rbase = blockIdx.x * ROWS;

  const v8f vzero = {0.f, 0.f, 0.f, 0.f, 0.f, 0.f, 0.f, 0.f};

  /* prefetch first weight tile immediately (overlaps the MLP prologue) */
  prefetch_w(0, wres_h, wf_h, wbuf0);

  /* ---- initial conditioning MLPs (wave-local: lane<16 owns one row) ---- */
  if (lane < 16){
    int mloc = m0 + lane;
    int r = rbase + mloc;
    float sg = sigmag[r];
    float a0 = n1b2[0], a1 = n1b2[1];
    #pragma unroll 4
    for (int j = 0; j < 32; ++j){
      float hp = n1w1[j] * sg + n1b1[j];
      float hs = hp / (1.f + expf(-hp));        /* silu */
      a0 += n1w2[j] * hs;
      a1 += n1w2[32 + j] * hs;
    }
    float i0 = z0g[2*r], i1 = z0g[2*r+1], i2 = xg[2*r], i3 = xg[2*r+1];
    float b0v = n2b2[0], b1v = n2b2[1];
    #pragma unroll 4
    for (int j = 0; j < 32; ++j){
      float hp = n2b1[j] + n2w1[j*4+0]*i0 + n2w1[j*4+1]*i1
                         + n2w1[j*4+2]*i2 + n2w1[j*4+3]*i3;
      float hs = hp / (1.f + expf(-hp));
      b0v += n2w2[j] * hs;
      b1v += n2w2[32 + j] * hs;
    }
    zbuf[mloc*2+0] = a0 + b0v;
    zbuf[mloc*2+1] = a1 + b1v;
    ldtot[mloc] = 0.f;
  }

  int t = 0;   /* global GEMM index; weight buffer = wbuf0 + (t&1)*WBUF_ELEMS */

  for (int li = 0; li < NL; ++li){
    const float* W0p = made_W0 + (size_t)li * HDIM * 2;
    const float* b0p = made_b0 + (size_t)li * HDIM;
    const float* bfp = made_bf + (size_t)li * (2 * PM);

    outb[m0 * 2 + lane] = 0.f;                  /* out = 0 (wave-local, 32 entries) */

    for (int pass = 0; pass < 2; ++pass){
      /* h = out[:,0]*W0[:,0] + b0 ; actA = relu(h)  (wave-local rows) */
      for (int idx = lane; idx < 16 * HDIM; idx += 32){
        int m = m0 + (idx >> 7), n = idx & 127;
        float v = outb[m * 2] * W0p[n * 2] + b0p[n];
        hbuf[m * HS + n] = v;
        actA[m * AS + n] = (half_t)fmaxf(v, 0.f);
      }

      for (int blk = 0; blk < NBLK; ++blk){
        /* ---- GEMM 2*blk : t1 = relu(relu(h) @ W0.T + b0) ---- */
        WAITBAR();
        prefetch_w(t + 1, wres_h, wf_h, wbuf0);
        v8f acc[8];
        #pragma unroll
        for (int nt = 0; nt < 8; ++nt) acc[nt] = vzero;
        gemm16<8>(&actA[m0 * AS], wbuf0 + (size_t)(t & 1) * WBUF_ELEMS, acc);
        {
          const float* brp = made_bres + (((size_t)li * NBLK + blk) * 2 + 0) * HDIM;
          #pragma unroll
          for (int nt = 0; nt < 8; ++nt){
            int n = nt * 16 + l15;
            float bv = brp[n];
            #pragma unroll
            for (int j = 0; j < 8; ++j){
              int m = m0 + hi * 8 + j;
              actA[m * AS + n] = (half_t)fmaxf(acc[nt][j] + bv, 0.f);
            }
          }
        }
        ++t;

        /* ---- GEMM 2*blk+1 : h += t1 @ W1.T + b1 ; refill actA ---- */
        WAITBAR();
        prefetch_w(t + 1, wres_h, wf_h, wbuf0);
        #pragma unroll
        for (int nt = 0; nt < 8; ++nt) acc[nt] = vzero;
        gemm16<8>(&actA[m0 * AS], wbuf0 + (size_t)(t & 1) * WBUF_ELEMS, acc);
        {
          const float* brp = made_bres + (((size_t)li * NBLK + blk) * 2 + 1) * HDIM;
          #pragma unroll
          for (int nt = 0; nt < 8; ++nt){
            int n = nt * 16 + l15;
            float bv = brp[n];
            #pragma unroll
            for (int j = 0; j < 8; ++j){
              int m = m0 + hi * 8 + j;
              hbuf[m * HS + n] += acc[nt][j] + bv;
            }
          }
        }
        /* actA = relu(h) for next block, or plain h copy before final proj */
        if (blk == 0){
          for (int idx = lane; idx < 16 * HDIM; idx += 32){
            int m = m0 + (idx >> 7), n = idx & 127;
            actA[m * AS + n] = (half_t)fmaxf(hbuf[m * HS + n], 0.f);
          }
        } else {
          for (int idx = lane; idx < 16 * HDIM; idx += 32){
            int m = m0 + (idx >> 7), n = idx & 127;
            actA[m * AS + n] = (half_t)hbuf[m * HS + n];
          }
        }
        ++t;
      }

      /* ---- final projection: p(dim=1) = h @ Wf[PM:2PM].T + bf[PM:] ---- */
      WAITBAR();
      prefetch_w(t + 1, wres_h, wf_h, wbuf0);
      {
        v8f accf[2]; accf[0] = vzero; accf[1] = vzero;
        gemm16<2>(&actA[m0 * AS], wbuf0 + (size_t)(t & 1) * WBUF_ELEMS, accf);
        #pragma unroll
        for (int nt = 0; nt < 2; ++nt){
          int n = nt * 16 + l15;
          if (n < PM){
            float bv = bfp[PM + n];
            #pragma unroll
            for (int j = 0; j < 8; ++j){
              int m = m0 + hi * 8 + j;
              pbuf[m * PS + n] = accf[nt][j] + bv;
            }
          }
        }
      }
      ++t;

      /* ---- RQS inverse spline: one lane per (row, dim), wave-local ---- */
      {
        const int mrow = m0 + l15;
        const int dsel = hi;
        float pp[PM];
        if (dsel == 0){
          #pragma unroll
          for (int k = 0; k < PM; ++k) pp[k] = bfp[k];   /* masked weights -> bias only */
        } else {
          #pragma unroll
          for (int k = 0; k < PM; ++k) pp[k] = pbuf[mrow * PS + k];
        }
        float y = zbuf[mrow * 2 + dsel];
        const float invs = 1.0f / SQH;

        float cw[9], chh[9];
        { /* widths cumulative (x domain) */
          float mx = -1e30f;
          #pragma unroll
          for (int k = 0; k < 8; ++k) mx = fmaxf(mx, pp[k] * invs);
          float e[8], s = 0.f;
          #pragma unroll
          for (int k = 0; k < 8; ++k){ e[k] = expf(pp[k] * invs - mx); s += e[k]; }
          float inv_s = 1.f / s, acc = 0.f;
          cw[0] = -TBOUND;
          #pragma unroll
          for (int k = 0; k < 8; ++k){
            acc += MINW + (1.f - MINW * NB) * e[k] * inv_s;
            cw[k + 1] = 2.f * TBOUND * acc - TBOUND;
          }
          cw[8] = TBOUND;
        }
        { /* heights cumulative (y domain) */
          float mx = -1e30f;
          #pragma unroll
          for (int k = 0; k < 8; ++k) mx = fmaxf(mx, pp[8 + k] * invs);
          float e[8], s = 0.f;
          #pragma unroll
          for (int k = 0; k < 8; ++k){ e[k] = expf(pp[8 + k] * invs - mx); s += e[k]; }
          float inv_s = 1.f / s, acc = 0.f;
          chh[0] = -TBOUND;
          #pragma unroll
          for (int k = 0; k < 8; ++k){
            acc += MINW + (1.f - MINW * NB) * e[k] * inv_s;
            chh[k + 1] = 2.f * TBOUND * acc - TBOUND;
          }
          chh[8] = TBOUND;
        }
        float dv[9];
        dv[0] = MIND + softplusf(DCONST);
        dv[8] = dv[0];
        #pragma unroll
        for (int k = 1; k < 8; ++k) dv[k] = MIND + softplusf(pp[15 + k]);

        bool inside = (y >= -TBOUND) && (y <= TBOUND);
        float yc = fminf(fmaxf(y, -TBOUND), TBOUND);
        int idx = 0;
        #pragma unroll
        for (int k = 1; k < 8; ++k) if (yc >= chh[k]) idx = k;
        /* register-resident bin select (no scratch for dynamic indexing) */
        float yk = chh[0], hk = chh[1]-chh[0], xk = cw[0], wk = cw[1]-cw[0];
        float dk = dv[0], dk1 = dv[1];
        #pragma unroll
        for (int k = 1; k < 8; ++k){
          bool sel = (idx == k);
          yk  = sel ? chh[k]          : yk;
          hk  = sel ? chh[k+1]-chh[k] : hk;
          xk  = sel ? cw[k]           : xk;
          wk  = sel ? cw[k+1]-cw[k]   : wk;
          dk  = sel ? dv[k]           : dk;
          dk1 = sel ? dv[k+1]         : dk1;
        }
        float sk = hk / wk;
        float dy = yc - yk;
        float t2 = dk + dk1 - 2.f * sk;
        float aq = dy * t2 + hk * (sk - dk);
        float bq = hk * dk - dy * t2;
        float cq = -sk * dy;
        float disc  = fmaxf(bq * bq - 4.f * aq * cq, 0.f);
        float theta = 2.f * cq / (-bq - sqrtf(disc));
        float xo = theta * wk + xk;
        float om = 1.f - theta;
        float denom = sk + t2 * theta * om;
        float dnum  = sk * sk * (dk1 * theta * theta + 2.f * sk * theta * om + dk * om * om);
        float ldf = logf(dnum) - 2.f * logf(denom);
        outb[mrow * 2 + dsel] = inside ? xo : y;
        ldb [mrow * 2 + dsel] = inside ? -ldf : 0.f;
      }
      /* no barrier: spline outputs are wave-local, DS ops in-order per wave */
    } /* pass */

    /* ---- LU inverse, permutation, logdet (wave-local: lane<16 per row) ---- */
    if (lane < 16){
      int m = m0 + lane;
      float d0 = softplusf(lu_diag[li*2+0]) + LU_EPS;
      float d1 = softplusf(lu_diag[li*2+1]) + LU_EPS;
      float l = lu_lower[li], u = lu_upper[li];
      float invdet = 1.f / (d0 * d1);
      float i00 = (l*u + d1) * invdet, i01 = -u * invdet;
      float i10 = -l * d0 * invdet,    i11 = d0 * invdet;
      float o0 = outb[m*2+0] - lu_bias[li*2+0];
      float o1 = outb[m*2+1] - lu_bias[li*2+1];
      float zn0 = i00 * o0 + i01 * o1;
      float zn1 = i10 * o0 + i11 * o1;
      int p0 = perms[li*2+0], p1 = perms[li*2+1];
      zbuf[m*2+0] = p0 ? zn1 : zn0;
      zbuf[m*2+1] = p1 ? zn1 : zn0;
      ldtot[m] += ldb[m*2+0] + ldb[m*2+1] - (logf(d0) + logf(d1));
    }
  } /* layer */

  if (lane < 16){
    int m = m0 + lane;
    int r = rbase + m;
    outp[2*r+0] = zbuf[m*2+0];
    outp[2*r+1] = zbuf[m*2+1];
    outp[(size_t)2*B + r] = ldtot[m];
  }
}

/* ---------------- host launcher ---------------- */

extern "C" void kernel_launch(void* const* d_in, const int* in_sizes, int n_in,
                              void* d_out, int out_size, void* d_ws, size_t ws_size,
                              hipStream_t stream) {
  (void)n_in; (void)out_size; (void)ws_size;
  const float* z0    = (const float*)d_in[0];
  const float* x     = (const float*)d_in[1];
  const float* sigma = (const float*)d_in[2];
  const float* n1w1  = (const float*)d_in[3];
  const float* n1b1  = (const float*)d_in[4];
  const float* n1w2  = (const float*)d_in[5];
  const float* n1b2  = (const float*)d_in[6];
  const float* n2w1  = (const float*)d_in[7];
  const float* n2b1  = (const float*)d_in[8];
  const float* n2w2  = (const float*)d_in[9];
  const float* n2b2  = (const float*)d_in[10];
  const float* made_W0   = (const float*)d_in[11];
  const float* made_b0   = (const float*)d_in[12];
  const float* made_Wres = (const float*)d_in[13];
  const float* made_bres = (const float*)d_in[14];
  const float* made_Wf   = (const float*)d_in[15];
  const float* made_bf   = (const float*)d_in[16];
  const float* lu_lower  = (const float*)d_in[17];
  const float* lu_upper  = (const float*)d_in[18];
  const float* lu_diag   = (const float*)d_in[19];
  const float* lu_bias   = (const float*)d_in[20];
  const int*   perms     = (const int*)d_in[21];

  const int B = in_sizes[0] / 2;

  half_t* wres_h = (half_t*)d_ws;
  half_t* wf_h   = wres_h + (size_t)NL * NBLK * 2 * HDIM * HDIM;

  int nres = NL * NBLK * 2 * HDIM * HDIM;
  cvt_wres<<<(nres + 255) / 256, 256, 0, stream>>>(made_Wres, wres_h, nres);
  int nwf = NL * 32 * HDIM;
  cvt_wf<<<(nwf + 255) / 256, 256, 0, stream>>>(made_Wf, wf_h);

  size_t shmem = (size_t)2 * WBUF_ELEMS * sizeof(half_t) /* wbuf x2 */
               + (size_t)ROWS * AS * sizeof(half_t)      /* actA    */
               + (size_t)ROWS * HS * sizeof(float)       /* hbuf    */
               + (size_t)ROWS * PS * sizeof(float)       /* pbuf    */
               + (size_t)ROWS * 2 * sizeof(float) * 3    /* zbuf/outb/ldb */
               + (size_t)ROWS * sizeof(float);           /* ldtot   */

  nsf_main<<<B / ROWS, NTHR, shmem, stream>>>(
      z0, x, sigma, n1w1, n1b1, n1w2, n1b2, n2w1, n2b1, n2w2, n2b2,
      made_W0, made_b0, wres_h, made_bres, wf_h, made_bf,
      lu_lower, lu_upper, lu_diag, lu_bias, perms, (float*)d_out, B);
}
